// SelfAttention_88923002896953
// MI455X (gfx1250) — matrix operand order repaired and checked by reading, not compiled, not run on hardware
//
#include <hip/hip_runtime.h>
#include <hip/hip_bf16.h>

// ---------------------------------------------------------------------------
// L2-distance multi-head self-attention for MI455X (gfx1250, wave32, WMMA).
// N=4096 tokens, C=512 channels, 8 heads x 64 dim.  All GEMMs + attention use
// v_wmma_f32_16x16x32_f16 (f16 inputs, f32 accumulate).  Working set fits in
// the 192MB L2, so fragments are loaded straight from global as b128s.
// Attention: flash-style online softmax over 64-key chunks, fully
// lane-parallel (shfl_xor butterflies in the C-fragment layout), per-lane
// deferred exp-sum accumulation; only the f16 P transpose goes through LDS.
// Workspace layout (needs ~22.2 MiB):
//   [0,4M)   X16   : tokens, f16, [4096][512]
//   [4M,6M)  W16   : {wq,wk,wv,wo} * gain, f16, 4x[512][512]
//   [6M,10M) Q16   : [4096][512]
//   [10M,14M)K16   : [4096][512]
//   [14M,18M)Vt16  : [512][4096]   (transposed so PV B-fragments are contiguous)
//   [18M,22M)O16   : [4096][512]
//   [22M,..) AA    : f32 [8][4096] row norms of q
// ---------------------------------------------------------------------------

typedef _Float16 h16v __attribute__((ext_vector_type(16)));
typedef _Float16 h8v  __attribute__((ext_vector_type(8)));
typedef float    f8v  __attribute__((ext_vector_type(8)));

#define NTOK   4096
#define CDIM   512
#define NHEAD  8
#define DHEAD  64
#define SCALE  0.125f                  // dhead^-0.5
#define GAIN   0.04419417382415922f    // 1/sqrt(512) (both gain_in and gain_out)

static __device__ __forceinline__ h16v join16(h8v a, h8v b) {
    return __builtin_shufflevector(a, b, 0,1,2,3,4,5,6,7,8,9,10,11,12,13,14,15);
}
static __device__ __forceinline__ h8v ld8(const _Float16* p) {
    return *reinterpret_cast<const h8v*>(p);
}
static __device__ __forceinline__ f8v wmma16(h16v a, h16v b, f8v c) {
    // (neg_a, A, neg_b, B, c_mod, C, reuse_a, reuse_b)
    return __builtin_amdgcn_wmma_f32_16x16x32_f16(false, a, false, b, (short)0, c, false, false);
}

// --------------------------- prep kernels ----------------------------------

__global__ void k_prep_x(const float* __restrict__ fmap, _Float16* __restrict__ X) {
    int i = blockIdx.x * 256 + threadIdx.x;        // i < 4096*512
    int n = i >> 9, c = i & 511;
    X[i] = (_Float16)fmap[c * NTOK + n];           // X[n][c] = fmap[c][n]
}

__global__ void k_prep_w(const float* __restrict__ wq, const float* __restrict__ wk,
                         const float* __restrict__ wv, const float* __restrict__ wo,
                         _Float16* __restrict__ W) {
    int i = blockIdx.x * 256 + threadIdx.x;        // i < 4*512*512
    int w = i >> 18, j = i & 262143;
    const float* src = (w == 0) ? wq : (w == 1) ? wk : (w == 2) ? wv : wo;
    W[i] = (_Float16)(src[j] * GAIN);
}

// --------------------------- QKV projection --------------------------------
// One wave per 16(n) x 64(o) strip of Q/K/V = X(4096x512) * W^T(512x512).
// Per K-step: 1 A fragment feeds 4 B fragments / 4 accumulators (4x A reuse).
// A frag (16x32 f16): lane<16 halves0-15 -> K+0..15 (contiguous, like B);
//                     lane>=16        -> K+16..31.
// B frag (32x16 f16): lane L = col, lanes<16 K=0..15, lanes>=16 K=16..31.

__global__ void __launch_bounds__(256) k_qkv(const _Float16* __restrict__ X,
                                             const _Float16* __restrict__ W,
                                             _Float16* __restrict__ Q,
                                             _Float16* __restrict__ K,
                                             _Float16* __restrict__ Vt) {
    int wid  = (blockIdx.x * 256 + threadIdx.x) >> 5;   // 3*256*8 = 6144 waves
    int lane = threadIdx.x & 31;
    int mat  = wid >> 11;                               // / 2048
    int t    = wid & 2047;
    int n0   = (t >> 3) << 4;                           // 16-row tile
    int og   = (t & 7) << 6;                            // 64-col o group
    int r16  = lane & 15, hi = lane >> 4;

    const _Float16* Arow = X + (n0 + r16) * CDIM;
    const _Float16* Wm   = W + mat * CDIM * CDIM;

    f8v acc[4] = {};
#pragma unroll 2
    for (int kk = 0; kk < CDIM; kk += 32) {
        h16v a = join16(ld8(Arow + kk + hi * 16), ld8(Arow + kk + hi * 16 + 8));
#pragma unroll
        for (int u = 0; u < 4; ++u) {
            const _Float16* Brow = Wm + (og + u * 16 + r16) * CDIM + kk + hi * 16;
            h16v b = join16(ld8(Brow), ld8(Brow + 8));
            acc[u] = wmma16(a, b, acc[u]);
        }
    }
#pragma unroll
    for (int u = 0; u < 4; ++u) {
#pragma unroll
        for (int r = 0; r < 8; ++r) {  // C frag: row = r + hi*8, col = r16
            int n = n0 + r + hi * 8;
            int o = og + u * 16 + r16;
            _Float16 v = (_Float16)acc[u][r];
            if (mat == 0)      Q[n * CDIM + o] = v;
            else if (mat == 1) K[n * CDIM + o] = v;
            else               Vt[o * NTOK + n] = v;      // transposed store
        }
    }
}

// --------------------------- q row norms -----------------------------------

__global__ void k_aa(const _Float16* __restrict__ Q, float* __restrict__ AA) {
    int i = blockIdx.x * 256 + threadIdx.x;        // i < 8*4096 : (head, n)
    int h = i >> 12, n = i & 4095;
    const _Float16* q = Q + n * CDIM + h * DHEAD;
    float s = 0.f;
#pragma unroll 8
    for (int d = 0; d < DHEAD; ++d) { float x = (float)q[d]; s += x * x; }
    AA[i] = s;
}

// --------------------------- attention -------------------------------------
// One single-wave workgroup per (head, 16 query rows).  64 chunks of 64 keys.
// sim = (2*qk - AA_i - AA_j) * scale (reference uses q norms for BOTH sides).
// Softmax is lane-parallel: in the C layout a row's 16 values share a VGPR
// index across a 16-lane half, so the row max is a 4-step shfl_xor butterfly
// (width 16).  The exp-sum is accumulated per-lane (each lane owns 4 columns
// per chunk) with the row-uniform corr factor, and reduced once at the end.
// Only the f16 P transpose uses LDS (double-buffered, 1 barrier/chunk;
// 1-wave workgroup => barrier ~ S_NOP).

__global__ void __launch_bounds__(32) k_attn(const _Float16* __restrict__ Q,
                                             const _Float16* __restrict__ K,
                                             const _Float16* __restrict__ Vt,
                                             const float* __restrict__ AA,
                                             _Float16* __restrict__ O) {
    __shared__ _Float16 pS[2][16 * 64];

    int head = blockIdx.x >> 8;
    int n0   = (blockIdx.x & 255) << 4;
    int lane = threadIdx.x & 31;
    int r16  = lane & 15, hi = lane >> 4;

    const float*    AAh  = AA + head * NTOK;
    const _Float16* Qrow = Q + (n0 + r16) * CDIM + head * DHEAD;
    const _Float16* Vth  = Vt + head * DHEAD * NTOK;

    // Q A-fragments for d-chunks [0,32) and [32,64): loaded once.
    h16v aq0 = join16(ld8(Qrow + hi * 16),      ld8(Qrow + hi * 16 + 8));
    h16v aq1 = join16(ld8(Qrow + 32 + hi * 16), ld8(Qrow + 32 + hi * 16 + 8));

    float aaq[8], m[8], lsum[8];
#pragma unroll
    for (int r = 0; r < 8; ++r) {
        aaq[r]  = AAh[n0 + r + hi * 8];
        m[r]    = -1e30f;
        lsum[r] = 0.f;       // per-lane partial exp-sum (this lane's columns)
    }

    f8v acc[4] = {};

    for (int j0 = 0; j0 < NTOK; j0 += 64) {
        int buf = (j0 >> 6) & 1;

        // ---- S = Q K^T for 16x64 chunk (4 tiles x 2 d-chunks = 8 WMMA) ----
        f8v s[4];
#pragma unroll
        for (int t = 0; t < 4; ++t) {
            const _Float16* Krow = K + (j0 + t * 16 + r16) * CDIM + head * DHEAD;
            h16v b0 = join16(ld8(Krow + hi * 16),      ld8(Krow + hi * 16 + 8));
            h16v b1 = join16(ld8(Krow + 32 + hi * 16), ld8(Krow + 32 + hi * 16 + 8));
            f8v c = {};
            c = wmma16(aq0, b0, c);
            s[t] = wmma16(aq1, b1, c);
        }
        float aak[4];
#pragma unroll
        for (int t = 0; t < 4; ++t) aak[t] = AAh[j0 + t * 16 + r16];

        // ---- lane-parallel online softmax ----
#pragma unroll
        for (int r = 0; r < 8; ++r) {
            float sim[4];
#pragma unroll
            for (int t = 0; t < 4; ++t)
                sim[t] = (2.f * s[t][r] - aaq[r] - aak[t]) * SCALE;
            // row max over 64 cols: local max of 4, then 16-lane butterfly
            float mx = fmaxf(fmaxf(sim[0], sim[1]), fmaxf(sim[2], sim[3]));
            mx = fmaxf(mx, __shfl_xor(mx, 1, 16));
            mx = fmaxf(mx, __shfl_xor(mx, 2, 16));
            mx = fmaxf(mx, __shfl_xor(mx, 4, 16));
            mx = fmaxf(mx, __shfl_xor(mx, 8, 16));
            mx = fmaxf(mx, m[r]);
            float corr = __expf(m[r] - mx);
            m[r] = mx;
            int row = (r + hi * 8) * 64;
            float ps = 0.f;
#pragma unroll
            for (int t = 0; t < 4; ++t) {
                float p = __expf(sim[t] - mx);
                pS[buf][row + t * 16 + r16] = (_Float16)p;
                ps += p;
            }
            lsum[r] = lsum[r] * corr + ps;     // deferred cross-lane reduce
            acc[0][r] *= corr; acc[1][r] *= corr;
            acc[2][r] *= corr; acc[3][r] *= corr;
        }
        __syncthreads();   // P transpose handoff (double-buffered LDS)

        // ---- O += P * V  (2 key-halves x 4 d-tiles = 8 WMMA) ----
#pragma unroll
        for (int kc = 0; kc < 2; ++kc) {
            h16v pa = join16(ld8(&pS[buf][r16 * 64 + kc * 32 + hi * 16]),
                             ld8(&pS[buf][r16 * 64 + kc * 32 + hi * 16 + 8]));
#pragma unroll
            for (int t = 0; t < 4; ++t) {
                const _Float16* Vrow =
                    Vth + (t * 16 + r16) * NTOK + j0 + kc * 32 + hi * 16;
                h16v bv = join16(ld8(Vrow), ld8(Vrow + 8));
                acc[t] = wmma16(pa, bv, acc[t]);
            }
        }
    }

    // ---- final: reduce per-lane partial sums, normalize, store ----
#pragma unroll
    for (int r = 0; r < 8; ++r) {
        float l = lsum[r];
        l += __shfl_xor(l, 1, 16);
        l += __shfl_xor(l, 2, 16);
        l += __shfl_xor(l, 4, 16);
        l += __shfl_xor(l, 8, 16);
        float inv = 1.0f / l;
        int n = n0 + r + hi * 8;
#pragma unroll
        for (int t = 0; t < 4; ++t)
            O[n * CDIM + head * DHEAD + t * 16 + r16] = (_Float16)(acc[t][r] * inv);
    }
}

// --------------------------- output projection -----------------------------
// out[o][n] = sum_c O16[n][c] * Wo16[o][c]   (f32 output, NCHW layout)
// Same 16x64-strip-per-wave structure as k_qkv.

__global__ void __launch_bounds__(256) k_out(const _Float16* __restrict__ O,
                                             const _Float16* __restrict__ Wo,
                                             float* __restrict__ out) {
    int wid  = (blockIdx.x * 256 + threadIdx.x) >> 5;   // 256*8 = 2048 waves
    int lane = threadIdx.x & 31;
    int n0   = (wid >> 3) << 4;
    int og   = (wid & 7) << 6;
    int r16  = lane & 15, hi = lane >> 4;

    const _Float16* Arow = O + (n0 + r16) * CDIM;

    f8v acc[4] = {};
#pragma unroll 2
    for (int kk = 0; kk < CDIM; kk += 32) {
        h16v a = join16(ld8(Arow + kk + hi * 16), ld8(Arow + kk + hi * 16 + 8));
#pragma unroll
        for (int u = 0; u < 4; ++u) {
            const _Float16* Brow = Wo + (og + u * 16 + r16) * CDIM + kk + hi * 16;
            h16v b = join16(ld8(Brow), ld8(Brow + 8));
            acc[u] = wmma16(a, b, acc[u]);
        }
    }
#pragma unroll
    for (int u = 0; u < 4; ++u) {
#pragma unroll
        for (int r = 0; r < 8; ++r) {
            int n = n0 + r + hi * 8;
            int o = og + u * 16 + r16;
            out[o * NTOK + n] = acc[u][r];
        }
    }
}

// --------------------------- launcher --------------------------------------

extern "C" void kernel_launch(void* const* d_in, const int* in_sizes, int n_in,
                              void* d_out, int out_size, void* d_ws, size_t ws_size,
                              hipStream_t stream) {
    const float* fmap = (const float*)d_in[0];
    const float* wq   = (const float*)d_in[1];
    const float* wk   = (const float*)d_in[2];
    const float* wv   = (const float*)d_in[3];
    const float* wo   = (const float*)d_in[4];

    char* ws = (char*)d_ws;
    _Float16* X16  = (_Float16*)(ws);
    _Float16* W16  = (_Float16*)(ws + (4u  << 20));
    _Float16* Q16  = (_Float16*)(ws + (6u  << 20));
    _Float16* K16  = (_Float16*)(ws + (10u << 20));
    _Float16* Vt16 = (_Float16*)(ws + (14u << 20));
    _Float16* O16  = (_Float16*)(ws + (18u << 20));
    float*    AA   = (float*)   (ws + (22u << 20));

    k_prep_x<<<(NTOK * CDIM) / 256, 256, 0, stream>>>(fmap, X16);
    k_prep_w<<<(4 * CDIM * CDIM) / 256, 256, 0, stream>>>(wq, wk, wv, wo, W16);
    // 3 mats * 256 n-tiles * 8 o-groups = 6144 waves, 8 waves/block
    k_qkv<<<768, 256, 0, stream>>>(X16, W16, Q16, K16, Vt16);
    k_aa<<<(NHEAD * NTOK) / 256, 256, 0, stream>>>(Q16, AA);
    // 8 heads * 256 query tiles, one wave per workgroup
    k_attn<<<NHEAD * 256, 32, 0, stream>>>(Q16, K16, Vt16, AA, O16);
    // 256 n-tiles * 8 o-groups = 2048 waves
    k_out<<<256, 256, 0, stream>>>(O16, W16 + 3 * CDIM * CDIM, (float*)d_out);
}
